// NSA_40793599378226
// MI455X (gfx1250) — compile-verified, matrix operand order checked by reading, not run on hardware
//
#include <hip/hip_runtime.h>
#include <hip/hip_bf16.h>

typedef _Float16 f16;
typedef __attribute__((ext_vector_type(16))) _Float16 v16h;
typedef __attribute__((ext_vector_type(8)))  float    v8f;

#define B_   4
#define T_   8192
#define D_   256
#define NB_  128
#define BS_  64
#define W_   512
#define TOPK_ 16
#define NEG_  (-1e30f)
#define SCALE_ 0.0625f   // 256^-0.5

// A-fragment (and Bᵀ-fragment) loader: element i of lane l holds
// (row = l&15, k = (i>>3)*16 + ((l>>4)&1)*8 + (i&7)), per ISA 7.12.2.
__device__ __forceinline__ v16h frag_kmajor(const f16* base, int ld, int lane) {
  const int m = lane & 15;
  const int h = (lane >> 4) & 1;
  v16h f;
#pragma unroll
  for (int i = 0; i < 16; ++i) {
    int k = ((i >> 3) << 4) + (h << 3) + (i & 7);
    f[i] = base[m * ld + k];
  }
  return f;
}

// B-fragment from a (K x N) row-major matrix: element (k-pattern, n = l&15).
__device__ __forceinline__ v16h frag_b_rowmajor(const f16* base, int ld, int lane) {
  const int n = lane & 15;
  const int h = (lane >> 4) & 1;
  v16h f;
#pragma unroll
  for (int i = 0; i < 16; ++i) {
    int k = ((i >> 3) << 4) + (h << 3) + (i & 7);
    f[i] = base[k * ld + n];
  }
  return f;
}

__device__ __forceinline__ v8f wmma_f16(v16h a, v16h b, v8f c) {
  return __builtin_amdgcn_wmma_f32_16x16x32_f16(false, a, false, b, (short)0, c,
                                                false, false);
}

// ---------------- Kernel 1: block-mean compression of K and V ----------------
__global__ void nsa_compress(const float* __restrict__ keys,
                             const float* __restrict__ values,
                             float* __restrict__ cK, float* __restrict__ cV,
                             f16* __restrict__ cKh, f16* __restrict__ cVh) {
  int blk = blockIdx.x;           // b*NB + n
  int d = threadIdx.x;            // 0..255 (one column per thread, coalesced)
  int b = blk / NB_, n = blk % NB_;
  const float* kp = keys   + ((size_t)(b * T_ + n * BS_)) * D_ + d;
  const float* vp = values + ((size_t)(b * T_ + n * BS_)) * D_ + d;
  float sk = 0.f, sv = 0.f;
#pragma unroll 4
  for (int r = 0; r < BS_; ++r) { sk += kp[(size_t)r * D_]; sv += vp[(size_t)r * D_]; }
  sk *= (1.0f / BS_); sv *= (1.0f / BS_);
  size_t o = (size_t)blk * D_ + d;
  cK[o] = sk; cV[o] = sv; cKh[o] = (f16)sk; cVh[o] = (f16)sv;
}

// ------------- Kernel 2: compressed + selective branches (fused) -------------
// One workgroup per (batch, 64-query tile); 256 threads = 8 waves.
__global__ void nsa_comp_sel(const float* __restrict__ q,
                             const float* __restrict__ gate_cmp,
                             const float* __restrict__ gate_slc,
                             const float* __restrict__ cV,
                             const f16* __restrict__ cKh,
                             const f16* __restrict__ cVh,
                             float* __restrict__ out) {
  extern __shared__ char smem[];
  f16*   Qh      = (f16*)smem;                 // 64*256 f16      : 32768 B
  float* S       = (float*)(smem + 32768);     // 64*128 f32      : 32768 B
  f16*   Pc      = (f16*)(smem + 65536);       // 64*128 f16      : 16384 B
  float* O       = (float*)(smem + 81920);     // 64*256 f32      : 65536 B
  float* sumFull = (float*)(smem + 147456);    // 64 f32          :   256 B
  int*   topIdx  = (int*)(smem + 147712);      // 64*16 i32       :  4096 B
  float* topVal  = (float*)(smem + 151808);    // 64*16 f32       :  4096 B
                                               // total 155904 B
  int b  = blockIdx.x >> 7;
  int qt = blockIdx.x & 127;
  int q0 = qt * BS_;
  int tid = threadIdx.x;
  int lane = tid & 31, wid = tid >> 5;

  // Q tile -> LDS f16 (coalesced)
  const float* qbase = q + ((size_t)(b * T_ + q0)) * D_;
  for (int idx = tid; idx < BS_ * D_; idx += 256) Qh[idx] = (f16)qbase[idx];
  __syncthreads();

  // S = scale * Q @ cK^T   (64x128), 32 tiles over 8 waves
  {
    int mt = wid & 3;
    int ng = wid >> 2;
    v16h a[8];
#pragma unroll
    for (int kk = 0; kk < 8; ++kk)
      a[kk] = frag_kmajor(Qh + (mt * 16) * D_ + kk * 32, D_, lane);
    const f16* ckb = cKh + (size_t)b * NB_ * D_;
#pragma unroll
    for (int j = 0; j < 4; ++j) {
      int nt = ng * 4 + j;
      v8f acc = {};
#pragma unroll
      for (int kk = 0; kk < 8; ++kk) {
        v16h bf = frag_kmajor(ckb + (nt * 16) * D_ + kk * 32, D_, lane);
        acc = wmma_f16(a[kk], bf, acc);
      }
      int h = (lane >> 4) & 1;
#pragma unroll
      for (int r = 0; r < 8; ++r)
        S[(mt * 16 + r + 8 * h) * NB_ + nt * 16 + (lane & 15)] = acc[r] * SCALE_;
    }
  }
  __syncthreads();

  // per-row: full softmax (comp), masked softmax + top-16 (selective)
  if (tid < BS_) {
    int row = tid;
    const float* srow = S + row * NB_;
    float m1 = srow[0];
    for (int n = 1; n < NB_; ++n) m1 = fmaxf(m1, srow[n]);
    float s1 = 0.f;
    for (int n = 0; n < NB_; ++n) {
      float e = __expf(srow[n] - m1);
      s1 += e;
      Pc[row * NB_ + n] = (f16)e;          // unnormalized; divide at the end
    }
    sumFull[row] = s1;

    float m2 = srow[0];
    for (int n = 1; n <= qt; ++n) m2 = fmaxf(m2, srow[n]);
    float* Pm = O + row * D_;              // reuse O as scratch (128 entries)
    float s2 = 0.f;
    for (int n = 0; n < NB_; ++n) {
      float e = (n <= qt) ? __expf(srow[n] - m2) : 0.f;
      Pm[n] = e; s2 += e;
    }
    float inv2 = 1.0f / s2;
    for (int t = 0; t < TOPK_; ++t) {
      float bv = -1.f; int bi = 0;
      for (int n = 0; n < NB_; ++n) { if (Pm[n] > bv) { bv = Pm[n]; bi = n; } }
      topIdx[row * TOPK_ + t] = bi;
      topVal[row * TOPK_ + t] = bv * inv2;
      Pm[bi] = -2.f;
    }
  }
  __syncthreads();

  // O = Pc @ cV   (64x256), 64 tiles over 8 waves
  {
    int mt = wid & 3;
    int ng = wid >> 2;
    v16h a[4];
#pragma unroll
    for (int kk = 0; kk < 4; ++kk)
      a[kk] = frag_kmajor(Pc + (mt * 16) * NB_ + kk * 32, NB_, lane);
    const f16* cvb = cVh + (size_t)b * NB_ * D_;
#pragma unroll
    for (int j = 0; j < 8; ++j) {
      int nt = ng * 8 + j;
      v8f acc = {};
#pragma unroll
      for (int kk = 0; kk < 4; ++kk) {
        v16h bf = frag_b_rowmajor(cvb + (kk * 32) * D_ + nt * 16, D_, lane);
        acc = wmma_f16(a[kk], bf, acc);
      }
      int h = (lane >> 4) & 1;
#pragma unroll
      for (int r = 0; r < 8; ++r)
        O[(mt * 16 + r + 8 * h) * D_ + nt * 16 + (lane & 15)] = acc[r];
    }
  }
  __syncthreads();

  // gated combine + selective gather-sum, single coalesced store
  const float* cvb = cV + (size_t)b * NB_ * D_;
  for (int idx = tid; idx < BS_ * D_; idx += 256) {
    int row = idx >> 8, d = idx & 255;
    int gi = b * T_ + q0 + row;
    float cval = O[idx] * (gate_cmp[gi] / sumFull[row]);
    float sval = 0.f;
#pragma unroll
    for (int t = 0; t < TOPK_; ++t)
      sval += topVal[row * TOPK_ + t] *
              cvb[(size_t)topIdx[row * TOPK_ + t] * D_ + d];
    out[(size_t)gi * D_ + d] = cval + sval * gate_slc[gi];
  }
}

// ------------- Kernel 3: sliding-window flash attention (W=512) -------------
// One workgroup per (batch, 64-query tile); 4 waves, each owns one 16-row tile
// and keeps its 16x256 f32 accumulator in VGPRs.
__global__ void nsa_sliding(const float* __restrict__ q,
                            const float* __restrict__ keys,
                            const float* __restrict__ values,
                            const float* __restrict__ gate_swa,
                            float* __restrict__ out) {
  extern __shared__ char smem[];
  f16* Qh = (f16*)smem;               // 64*256 f16 : 32768 B
  f16* Kh = (f16*)(smem + 32768);     // 64*256 f16 : 32768 B
  f16* Vh = (f16*)(smem + 65536);     // 64*256 f16 : 32768 B
  f16* Pb = (f16*)(smem + 98304);     // 4 waves * 16*64 f16 : 8192 B
  int b  = blockIdx.x >> 7;
  int qt = blockIdx.x & 127;
  int q0 = qt * BS_;
  int tid  = threadIdx.x;
  int lane = tid & 31, w = tid >> 5;  // 4 waves
  int h  = (lane >> 4) & 1;
  int ln = lane & 15;

  const float* qbase = q + ((size_t)(b * T_ + q0)) * D_;
  for (int idx = tid; idx < BS_ * D_; idx += 128) Qh[idx] = (f16)qbase[idx];
  __syncthreads();

  v16h qa[8];
#pragma unroll
  for (int kk = 0; kk < 8; ++kk)
    qa[kk] = frag_kmajor(Qh + (w * 16) * D_ + kk * 32, D_, lane);

  float mrow[8], lrow[8];
  v8f Oacc[16];
#pragma unroll
  for (int r = 0; r < 8; ++r) { mrow[r] = NEG_; lrow[r] = 0.f; }
#pragma unroll
  for (int nt = 0; nt < 16; ++nt) { v8f z = {}; Oacc[nt] = z; }

  f16* Pw = Pb + w * (16 * 64);

  for (int c = 0; c < 9; ++c) {
    int kc = q0 + (c - 8) * BS_;     // chunk start; multiple of 64
    if (kc < 0) continue;            // uniform across the block
    __syncthreads();                 // protect Kh/Vh from previous iteration
    const float* kb = keys   + ((size_t)(b * T_ + kc)) * D_;
    const float* vb = values + ((size_t)(b * T_ + kc)) * D_;
    for (int idx = tid; idx < BS_ * D_; idx += 128) {
      Kh[idx] = (f16)kb[idx];
      Vh[idx] = (f16)vb[idx];
    }
    __syncthreads();

    // S tile (16x64) for this wave's queries
    float s[4][8];
#pragma unroll
    for (int nt = 0; nt < 4; ++nt) {
      v8f acc = {};
#pragma unroll
      for (int kk = 0; kk < 8; ++kk) {
        v16h bf = frag_kmajor(Kh + (nt * 16) * D_ + kk * 32, D_, lane);
        acc = wmma_f16(qa[kk], bf, acc);
      }
      int kpos = kc + nt * 16 + ln;
#pragma unroll
      for (int r = 0; r < 8; ++r) {
        int qpos = q0 + w * 16 + r + 8 * h;
        bool ok = (kpos <= qpos) && (kpos + W_ > qpos);   // [qpos-511..qpos]
        s[nt][r] = ok ? acc[r] * SCALE_ : NEG_;
      }
    }

    // online softmax per row (row = C-layout VGPR slot; reduce across 16 lanes)
#pragma unroll
    for (int r = 0; r < 8; ++r) {
      float rm = s[0][r];
#pragma unroll
      for (int nt = 1; nt < 4; ++nt) rm = fmaxf(rm, s[nt][r]);
#pragma unroll
      for (int off = 1; off < 16; off <<= 1)
        rm = fmaxf(rm, __shfl_xor(rm, off, 16));
      float mnew = fmaxf(mrow[r], rm);
      float sf = __expf(mrow[r] - mnew);
      mrow[r] = mnew;
      float ps = 0.f;
#pragma unroll
      for (int nt = 0; nt < 4; ++nt) {
        float p = (s[nt][r] > -1e29f) ? __expf(s[nt][r] - mnew) : 0.f;
        s[nt][r] = p; ps += p;
      }
#pragma unroll
      for (int off = 1; off < 16; off <<= 1)
        ps += __shfl_xor(ps, off, 16);
      lrow[r] = lrow[r] * sf + ps;
#pragma unroll
      for (int nt = 0; nt < 16; ++nt) Oacc[nt][r] *= sf;
    }

    // C-layout -> A-layout for P via per-wave LDS staging
#pragma unroll
    for (int nt = 0; nt < 4; ++nt)
#pragma unroll
      for (int r = 0; r < 8; ++r)
        Pw[(r + 8 * h) * 64 + nt * 16 + ln] = (f16)s[nt][r];
    __syncthreads();

    // O += P @ V
#pragma unroll
    for (int nt = 0; nt < 16; ++nt) {
#pragma unroll
      for (int kk = 0; kk < 2; ++kk) {
        v16h pa = frag_kmajor(Pw + kk * 32, 64, lane);
        v16h bf = frag_b_rowmajor(Vh + (kk * 32) * D_ + nt * 16, D_, lane);
        Oacc[nt] = wmma_f16(pa, bf, Oacc[nt]);
      }
    }
  }

  // finalize: gated add into out (kernel 2 already wrote comp+sel)
#pragma unroll
  for (int r = 0; r < 8; ++r) {
    int qpos = q0 + w * 16 + r + 8 * h;
    int gi = b * T_ + qpos;
    float g = gate_swa[gi];
    float invl = (lrow[r] > 0.f) ? 1.0f / lrow[r] : 0.f;
#pragma unroll
    for (int nt = 0; nt < 16; ++nt) {
      size_t oi = (size_t)gi * D_ + nt * 16 + ln;
      out[oi] += Oacc[nt][r] * g * invl;
    }
  }
}

extern "C" void kernel_launch(void* const* d_in, const int* in_sizes, int n_in,
                              void* d_out, int out_size, void* d_ws, size_t ws_size,
                              hipStream_t stream) {
  (void)in_sizes; (void)n_in; (void)out_size; (void)ws_size;
  const float* q  = (const float*)d_in[0];
  const float* k  = (const float*)d_in[1];
  const float* v  = (const float*)d_in[2];
  const float* gc = (const float*)d_in[3];
  const float* gs = (const float*)d_in[4];
  const float* gw = (const float*)d_in[5];
  float* out = (float*)d_out;

  char* ws = (char*)d_ws;
  float* cK = (float*)ws;                                   // 512 KB
  float* cV = cK + (size_t)B_ * NB_ * D_;                   // 512 KB
  f16* cKh = (f16*)(cV + (size_t)B_ * NB_ * D_);            // 256 KB
  f16* cVh = cKh + (size_t)B_ * NB_ * D_;                   // 256 KB

  nsa_compress<<<B_ * NB_, 256, 0, stream>>>(k, v, cK, cV, cKh, cVh);

  size_t lds2 = 155904;
  nsa_comp_sel<<<B_ * (T_ / BS_), 256, lds2, stream>>>(q, gc, gs, cV, cKh, cVh, out);

  size_t lds3 = 98304 + 8192;
  nsa_sliding<<<B_ * (T_ / BS_), 128, lds3, stream>>>(q, k, v, gw, out);
}
